// Encoder_23476291240695
// MI455X (gfx1250) — matrix operand order repaired and checked by reading, not compiled, not run on hardware
//
#include <hip/hip_runtime.h>
#include <cstdint>

// ---------------------------------------------------------------------------
// Transformer encoder (L=6,H=8,D=512,DH=64,N=300,B=64,DFF=2048) for gfx1250.
// GEMMs: v_wmma_f32_16x16x32_bf16, double-buffered LDS.
// Tile staging: TDM tensor_load_to_lds (TENSORcnt) when available, else
// global_load_async_to_lds_b128 (ASYNCcnt), else plain copies.
// Weights converted once to bf16 TRANSPOSED [N,K] so A/B tiles are straight
// 2D row-major tiles (TDM-friendly) and fragments load contiguously.
// ---------------------------------------------------------------------------

typedef __bf16 bf16;
typedef __attribute__((ext_vector_type(16))) __bf16 v16bf;
typedef __attribute__((ext_vector_type(8)))  float  v8f;
typedef __attribute__((ext_vector_type(4)))  int    v4i;
typedef __attribute__((ext_vector_type(8)))  int    v8i;
typedef __attribute__((ext_vector_type(4)))  unsigned int v4u;

#define LL   6
#define HH   8
#define DD   512
#define DHH  64
#define NN   300
#define BB   64
#define DFF  2048
#define BN   (BB * NN)      // 19200
#define NPAD 320            // keys padded to multiple of 32

// ----------------------------- feature probes ------------------------------
#if defined(__has_builtin)
#if __has_builtin(__builtin_amdgcn_global_load_async_to_lds_b128)
#define HAVE_ASYNC_LDS 1
#endif
#if __has_builtin(__builtin_amdgcn_tensor_load_to_lds)
#define HAVE_TDM 1
#endif
#endif
#ifndef HAVE_ASYNC_LDS
#define HAVE_ASYNC_LDS 0
#endif
#ifndef HAVE_TDM
#define HAVE_TDM 0
#endif

#define AS1 __attribute__((address_space(1)))
#define AS3 __attribute__((address_space(3)))

// --------------------- per-lane async copy (ASYNCcnt) ----------------------
__device__ __forceinline__ void copy16_to_lds(const bf16* g, bf16* l) {
#if HAVE_ASYNC_LDS
    __builtin_amdgcn_global_load_async_to_lds_b128(
        (AS1 v4i*)(uintptr_t)g,
        (AS3 v4i*)(uint32_t)(uintptr_t)l, 0, 0);
#else
    *reinterpret_cast<uint4*>(l) = *reinterpret_cast<const uint4*>(g);
#endif
}

__device__ __forceinline__ void async_join() {
#if HAVE_ASYNC_LDS
#if __has_builtin(__builtin_amdgcn_s_wait_asynccnt)
    __builtin_amdgcn_s_wait_asynccnt(0);
#else
    asm volatile("s_wait_asynccnt 0x0" ::: "memory");
#endif
#endif
}

// --------------------------- TDM copy (TENSORcnt) --------------------------
#if HAVE_TDM
// 2D tile (tile_rows x tile_cols bf16 elements, row stride in elements)
// global -> LDS, packed row-major at the LDS destination.
__device__ __forceinline__ void tdm_load_2d(const bf16* gsrc, bf16* ldst,
                                            int tile_cols, int tile_rows,
                                            int stride_elems) {
    uint64_t ga = (uint64_t)(uintptr_t)gsrc;
    uint32_t la = (uint32_t)(uintptr_t)ldst;   // LDS offset = flat[31:0]
    v4u g0;
    g0[0] = 1u;                                   // count=1, user mode
    g0[1] = la;                                   // lds_addr [63:32]
    g0[2] = (uint32_t)ga;                         // global_addr [95:64]
    g0[3] = (uint32_t)((ga >> 32) & 0x01FFFFFFu)  // global_addr [120:96]
            | 0x80000000u;                        // type=2 ("image") [127:126]
    const uint32_t dim0 = 0x40000000u;            // tensor_dim0 (huge, no OOB)
    const uint32_t dim1 = 0x00100000u;            // tensor_dim1 (huge, no OOB)
    v8i g1;
    g1[0] = (int)(1u << 16);                      // data_size=1 (2B); mask=0
    g1[1] = (int)((dim0 & 0xFFFFu) << 16);        // dim0[15:0] -> bits 63:48
    g1[2] = (int)((dim0 >> 16) | ((dim1 & 0xFFFFu) << 16));
    g1[3] = (int)((dim1 >> 16) | ((uint32_t)tile_cols << 16)); // tile_dim0
    g1[4] = tile_rows;                            // tile_dim1 (tile_dim2=0)
    g1[5] = stride_elems;                         // tensor_dim0_stride[31:0]
    g1[6] = 0;                                    // stride hi / dim1_stride lo
    g1[7] = 0;
    v4i z4 = {0, 0, 0, 0};
    v8i z8 = {0, 0, 0, 0, 0, 0, 0, 0};
    __builtin_amdgcn_tensor_load_to_lds(g0, g1, z4, z4, z8, 0);
}

__device__ __forceinline__ void tdm_join() {
#if __has_builtin(__builtin_amdgcn_s_wait_tensorcnt)
    __builtin_amdgcn_s_wait_tensorcnt((short)0);
#else
    asm volatile("s_wait_tensorcnt 0x0" ::: "memory");
#endif
}
#endif  // HAVE_TDM

// ------------------------------- WMMA helpers ------------------------------
__device__ __forceinline__ v8f wmma_bf16(v16bf a, v16bf b, v8f c) {
    return __builtin_amdgcn_wmma_f32_16x16x32_bf16(
        false, a, false, b, (short)0, c, false, false);
}

// A fragment (16x32 bf16): lane row = lane&15; half = lane>>4 selects K run
// {half*16..+15} -> two contiguous 16B LDS loads.
__device__ __forceinline__ v16bf load_frag_a(const bf16* row, int half) {
    union { v16bf v; uint4 q[2]; } u;
    u.q[0] = *reinterpret_cast<const uint4*>(row + half * 16);
    u.q[1] = *reinterpret_cast<const uint4*>(row + half * 16 + 8);
    return u.v;
}

// B fragment (32x16 bf16): lane col = lane&15, K = half*16 + j contiguous.
__device__ __forceinline__ v16bf load_frag_b16(const bf16* p) {
    union { v16bf v; uint4 q[2]; } u;
    u.q[0] = *reinterpret_cast<const uint4*>(p);
    u.q[1] = *reinterpret_cast<const uint4*>(p + 8);
    return u.v;
}

__device__ __forceinline__ v16bf load_frag_b_strided(const bf16* p, int stride) {
    union { v16bf v; bf16 e[16]; } u;
#pragma unroll
    for (int j = 0; j < 16; ++j) u.e[j] = p[j * stride];
    return u.v;
}

__device__ __forceinline__ v16bf load_frag_a_f32(const float* row, int half) {
    union { v16bf v; bf16 e[16]; } u;
#pragma unroll
    for (int i = 0; i < 8; ++i) u.e[i]     = (bf16)row[half * 16 + i];
#pragma unroll
    for (int i = 0; i < 8; ++i) u.e[8 + i] = (bf16)row[half * 16 + 8 + i];
    return u.v;
}

// ---------------------------------------------------------------------------
// Weight conversion (one-time): fp32 -> bf16, transposed per layer.
// src [L][R][C] -> dst [L][C][R]
// ---------------------------------------------------------------------------
__global__ void cvt_transpose_kernel(const float* __restrict__ src,
                                     bf16* __restrict__ dst,
                                     int R, int C, size_t total) {
    size_t i = (size_t)blockIdx.x * blockDim.x + threadIdx.x;
    if (i >= total) return;
    int r = (int)(i % (size_t)R);
    size_t t = i / (size_t)R;
    int c = (int)(t % (size_t)C);
    int l = (int)(t / (size_t)C);
    dst[i] = (bf16)src[((size_t)l * R + r) * C + c];
}

// Wk [L,H,D,DH] -> dst [L][n=h*64+e][d]  (transposed projection matrix)
__global__ void cvt_wkT_kernel(const float* __restrict__ wk,
                               bf16* __restrict__ dst) {
    size_t i = (size_t)blockIdx.x * blockDim.x + threadIdx.x;
    if (i >= (size_t)LL * DD * DD) return;
    int d = (int)(i & (DD - 1));
    size_t r = i >> 9;
    int n = (int)(r & (DD - 1));
    int l = (int)(r >> 9);
    int h = n >> 6, e = n & 63;
    dst[i] = (bf16)wk[(((size_t)l * HH + h) * DD + d) * DHH + e];
}

// ---------------------------------------------------------------------------
// Embedding: x = emb[tok] + pos ; fp32 + bf16 copies
// ---------------------------------------------------------------------------
__global__ void embed_kernel(const int* __restrict__ tok,
                             const float* __restrict__ emb,
                             const float* __restrict__ pos,
                             float* __restrict__ xF, bf16* __restrict__ xB) {
    size_t idx = (size_t)blockIdx.x * blockDim.x + threadIdx.x;
    if (idx >= (size_t)BN * DD) return;
    int d = (int)(idx & (DD - 1));
    size_t rn = idx >> 9;
    int n = (int)(rn % NN);
    int t = tok[rn];
    float v = emb[(size_t)t * DD + d] + pos[(size_t)n * DD + d];
    xF[idx] = v;
    xB[idx] = (bf16)v;
}

// ---------------------------------------------------------------------------
// bf16 WMMA GEMM: C[M,N] = A[M,K] * Bt[N,K]^T  (+bias +residual +relu)
// 64x64 block tile, 4 waves. Double-buffered LDS; tiles staged by TDM
// (wave 0 issues, TENSORcnt) or per-lane async loads.
// ---------------------------------------------------------------------------
template <int HAS_BIAS, int HAS_RES, int RELU, int OUTF, int OUTB>
__global__ __launch_bounds__(128) void gemm_bf16_kernel(
    const bf16* __restrict__ A, const bf16* __restrict__ Bt,
    int M, int N, int K,
    const float* __restrict__ bias, const float* __restrict__ resid,
    float* __restrict__ outF, bf16* __restrict__ outB) {
    __shared__ bf16 As[2][64][32];
    __shared__ bf16 Bs[2][64][32];   // holds Bt rows: [n][k]

    const int m0   = blockIdx.y * 64;
    const int n0   = blockIdx.x * 64;
    const int tid  = threadIdx.x;
    const int wave = tid >> 5;
    const int lane = tid & 31;
    const int col  = lane & 15;
    const int half = lane >> 4;

    v8f acc[4] = {v8f{0}, v8f{0}, v8f{0}, v8f{0}};
    const int nsteps = K >> 5;

#if HAVE_TDM
    const bf16* aTile = A  + (size_t)m0 * K;
    const bf16* bTile = Bt + (size_t)n0 * K;
    if (wave == 0) {
        tdm_load_2d(aTile, &As[0][0][0], 32, 64, K);
        tdm_load_2d(bTile, &Bs[0][0][0], 32, 64, K);
        tdm_join();
    }
    __syncthreads();
    for (int s = 0; s < nsteps; ++s) {
        const int buf = s & 1;
        if (wave == 0 && s + 1 < nsteps) {   // TDM prefetch of next tile
            int k0 = (s + 1) << 5;
            tdm_load_2d(aTile + k0, &As[buf ^ 1][0][0], 32, 64, K);
            tdm_load_2d(bTile + k0, &Bs[buf ^ 1][0][0], 32, 64, K);
        }
        v16bf af = load_frag_a(&As[buf][wave * 16 + col][0], half);
#pragma unroll
        for (int t = 0; t < 4; ++t) {
            v16bf bfr = load_frag_b16(&Bs[buf][t * 16 + col][half * 16]);
            acc[t] = wmma_bf16(af, bfr, acc[t]);
        }
        if (wave == 0) tdm_join();
        __syncthreads();
    }
#else
    // staging map: thread copies 16B of one A row and 16B of one Bt row
    const int  mrow = tid >> 1;          // 0..63
    const int  kh   = (tid & 1) * 16;    // 0 or 16
    const bf16* aSrc = A  + (size_t)(m0 + mrow) * K + kh;
    const bf16* bSrc = Bt + (size_t)(n0 + mrow) * K + kh;

    copy16_to_lds(aSrc, &As[0][mrow][kh]);
    copy16_to_lds(bSrc, &Bs[0][mrow][kh]);
    async_join();
    __syncthreads();
    for (int s = 0; s < nsteps; ++s) {
        const int buf = s & 1;
        if (s + 1 < nsteps) {
            int k0 = (s + 1) << 5;
            copy16_to_lds(aSrc + k0, &As[buf ^ 1][mrow][kh]);
            copy16_to_lds(bSrc + k0, &Bs[buf ^ 1][mrow][kh]);
        }
        v16bf af = load_frag_a(&As[buf][wave * 16 + col][0], half);
#pragma unroll
        for (int t = 0; t < 4; ++t) {
            v16bf bfr = load_frag_b16(&Bs[buf][t * 16 + col][half * 16]);
            acc[t] = wmma_bf16(af, bfr, acc[t]);
        }
        async_join();
        __syncthreads();
    }
#endif

    const int rbase = m0 + wave * 16 + half * 8;
#pragma unroll
    for (int t = 0; t < 4; ++t) {
        int c = n0 + t * 16 + col;
        float bv = HAS_BIAS ? bias[c] : 0.0f;
#pragma unroll
        for (int i = 0; i < 8; ++i) {
            size_t r = (size_t)(rbase + i);
            float v = acc[t][i] + bv;
            if (HAS_RES) v += resid[r * N + c];
            if (RELU)    v = v > 0.0f ? v : 0.0f;
            if (OUTF)    outF[r * N + c] = v;
            if (OUTB)    outB[r * N + c] = (bf16)v;
        }
    }
}

// ---------------------------------------------------------------------------
// Fused attention for one (b, h, 16-row tile). Q=K=V (shared projection).
// K tile [320x64] bf16 + fp32 score strip [16x320] in LDS; WMMA both GEMMs.
// K tile arrives via one TDM 2D transfer (300x64, stride 512) when available.
// ---------------------------------------------------------------------------
__global__ __launch_bounds__(32) void attn_kernel(const bf16* __restrict__ kbuf,
                                                  bf16* __restrict__ obuf) {
    __shared__ bf16  Ks[NPAD][DHH];   // 40 KB
    __shared__ float Ss[16][NPAD];    // 20 KB
    __shared__ float invs[16];

    const int r0   = blockIdx.x * 16;
    const int h    = blockIdx.y;
    const int b    = blockIdx.z;
    const int lane = threadIdx.x;
    const int col  = lane & 15;
    const int half = lane >> 4;

    const bf16* kb = kbuf + (size_t)b * NN * DD + h * DHH;

#if HAVE_TDM
    tdm_load_2d(kb, &Ks[0][0], DHH, NN, DD);   // one 300x64 tile -> LDS
#else
    for (int r = lane; r < NN; r += 32) {
        const bf16* src = kb + (size_t)r * DD;
#pragma unroll
        for (int j = 0; j < 8; ++j)
            copy16_to_lds(src + j * 8, &Ks[r][j * 8]);
    }
#endif
    if (lane < NPAD - NN) {            // zero-pad rows 300..319
        uint4 z = {0u, 0u, 0u, 0u};
        uint4* dst = reinterpret_cast<uint4*>(&Ks[NN + lane][0]);
#pragma unroll
        for (int j = 0; j < 8; ++j) dst[j] = z;
    }
#if HAVE_TDM
    tdm_join();
#else
    async_join();
#endif
    __syncthreads();

    v16bf qa0 = load_frag_a(&Ks[r0 + col][0], half);
    v16bf qa1 = load_frag_a(&Ks[r0 + col][32], half);

    const float scale = 0.04419417382415922f;  // 1/sqrt(512)

    // S = Q * K^T (scaled)
    for (int c = 0; c < NPAD / 16; ++c) {
        v8f acc = v8f{0};
        v16bf b0 = load_frag_b16(&Ks[c * 16 + col][half * 16]);
        acc = wmma_bf16(qa0, b0, acc);
        v16bf b1 = load_frag_b16(&Ks[c * 16 + col][32 + half * 16]);
        acc = wmma_bf16(qa1, b1, acc);
#pragma unroll
        for (int i = 0; i < 8; ++i)
            Ss[half * 8 + i][c * 16 + col] = acc[i] * scale;
    }
    __syncthreads();

    // Mask padded key columns
    for (int idx = lane; idx < 16 * (NPAD - NN); idx += 32) {
        int r = idx / (NPAD - NN);
        int c = NN + idx % (NPAD - NN);
        Ss[r][c] = -1e30f;
    }
    __syncthreads();

    // Softmax: lane pair (l, l+16) shares row = lane&15 (even/odd columns)
    {
        int row = lane & 15;
        float mx = -1e30f;
        for (int c = half; c < NPAD; c += 2) mx = fmaxf(mx, Ss[row][c]);
        mx = fmaxf(mx, __shfl_xor(mx, 16, 32));
        float sum = 0.0f;
        for (int c = half; c < NPAD; c += 2) {
            float e = __expf(Ss[row][c] - mx);
            Ss[row][c] = e;
            sum += e;
        }
        sum += __shfl_xor(sum, 16, 32);
        if (half == 0) invs[row] = 1.0f / sum;
    }
    __syncthreads();

    // O = P * K  (P fragments built from fp32 LDS, 1/sum folded in epilogue)
    bf16* ob = obuf + ((size_t)b * NN + r0) * DD + h * DHH;
#pragma unroll
    for (int t = 0; t < 4; ++t) {
        v8f acc = v8f{0};
        for (int kk = 0; kk < NPAD; kk += 32) {
            v16bf a  = load_frag_a_f32(&Ss[col][kk], half);
            v16bf bb = load_frag_b_strided(&Ks[kk + half * 16][t * 16 + col], DHH);
            acc = wmma_bf16(a, bb, acc);
        }
#pragma unroll
        for (int i = 0; i < 8; ++i) {
            int r = half * 8 + i;
            if (r0 + r < NN)
                ob[(size_t)r * DD + t * 16 + col] = (bf16)(acc[i] * invs[r]);
        }
    }
}

// ---------------------------------------------------------------------------
// LayerNorm: one wave per row (D=512 -> 16 elems/lane), shuffle reductions
// ---------------------------------------------------------------------------
__global__ __launch_bounds__(256) void layernorm_kernel(
    const float* __restrict__ in, const float* __restrict__ gamma,
    const float* __restrict__ beta, float* __restrict__ outF,
    bf16* __restrict__ outB, int rows) {
    int wave = threadIdx.x >> 5, lane = threadIdx.x & 31;
    int row = blockIdx.x * 8 + wave;
    if (row >= rows) return;
    const float* x = in + (size_t)row * DD;
    float v[16];
    float s = 0.0f;
#pragma unroll
    for (int i = 0; i < 16; ++i) { v[i] = x[lane + i * 32]; s += v[i]; }
#pragma unroll
    for (int off = 16; off >= 1; off >>= 1) s += __shfl_xor(s, off, 32);
    float mean = s * (1.0f / DD);
    float var = 0.0f;
#pragma unroll
    for (int i = 0; i < 16; ++i) { float d = v[i] - mean; var += d * d; }
#pragma unroll
    for (int off = 16; off >= 1; off >>= 1) var += __shfl_xor(var, off, 32);
    float inv = rsqrtf(var * (1.0f / DD) + 1e-5f);
    float* oF = outF + (size_t)row * DD;
    bf16*  oB = outB + (size_t)row * DD;
#pragma unroll
    for (int i = 0; i < 16; ++i) {
        int d = lane + i * 32;
        float y = (v[i] - mean) * inv * gamma[d] + beta[d];
        oF[d] = y;
        oB[d] = (bf16)y;
    }
}

// ---------------------------------------------------------------------------
// Host orchestration
// ---------------------------------------------------------------------------
extern "C" void kernel_launch(void* const* d_in, const int* in_sizes, int n_in,
                              void* d_out, int out_size, void* d_ws, size_t ws_size,
                              hipStream_t stream) {
    const int*   tokens = (const int*)  d_in[0];
    const float* emb    = (const float*)d_in[1];
    const float* pos    = (const float*)d_in[2];
    const float* Wk     = (const float*)d_in[3];
    const float* Wo     = (const float*)d_in[4];
    const float* W1     = (const float*)d_in[5];
    const float* b1     = (const float*)d_in[6];
    const float* W2     = (const float*)d_in[7];
    const float* b2     = (const float*)d_in[8];
    const float* gamma  = (const float*)d_in[9];
    const float* beta   = (const float*)d_in[10];

    char* ws = (char*)d_ws;
    size_t off = 0;
    auto alloc = [&](size_t bytes) -> void* {
        void* p = ws + off;
        off += (bytes + 255) & ~(size_t)255;
        return p;
    };

    // transposed bf16 weights: [N][K] per layer
    bf16*  wkT  = (bf16*) alloc((size_t)LL * DD * DD * 2);
    bf16*  woT  = (bf16*) alloc((size_t)LL * DD * DD * 2);
    bf16*  w1T  = (bf16*) alloc((size_t)LL * DFF * DD * 2);
    bf16*  w2T  = (bf16*) alloc((size_t)LL * DD * DFF * 2);
    float* xF   = (float*)alloc((size_t)BN * DD * 4);
    bf16*  xB   = (bf16*) alloc((size_t)BN * DD * 2);
    bf16*  kB   = (bf16*) alloc((size_t)BN * DD * 2);
    bf16*  oB   = (bf16*) alloc((size_t)BN * DD * 2);
    float* preF = (float*)alloc((size_t)BN * DD * 4);
    float* zF   = (float*)alloc((size_t)BN * DD * 4);
    bf16*  zB   = (bf16*) alloc((size_t)BN * DD * 2);
    bf16*  hB   = (bf16*) alloc((size_t)BN * DFF * 2);

    // --- one-time transposed weight conversion ---
    {
        size_t n = (size_t)LL * DD * DD;
        cvt_wkT_kernel<<<(unsigned)((n + 255) / 256), 256, 0, stream>>>(Wk, wkT);
        cvt_transpose_kernel<<<(unsigned)((n + 255) / 256), 256, 0, stream>>>(
            Wo, woT, DD, DD, n);                     // [din][dout] -> [dout][din]
        size_t n2 = (size_t)LL * DD * DFF;
        cvt_transpose_kernel<<<(unsigned)((n2 + 255) / 256), 256, 0, stream>>>(
            W1, w1T, DD, DFF, n2);                   // [d][f] -> [f][d]
        cvt_transpose_kernel<<<(unsigned)((n2 + 255) / 256), 256, 0, stream>>>(
            W2, w2T, DFF, DD, n2);                   // [f][d] -> [d][f]
    }

    // --- embedding ---
    {
        size_t n = (size_t)BN * DD;
        embed_kernel<<<(unsigned)((n + 255) / 256), 256, 0, stream>>>(
            tokens, emb, pos, xF, xB);
    }

    for (int l = 0; l < LL; ++l) {
        // k = x @ Wk[l]  -> bf16
        gemm_bf16_kernel<0, 0, 0, 0, 1><<<dim3(DD / 64, BN / 64), 128, 0, stream>>>(
            xB, wkT + (size_t)l * DD * DD, BN, DD, DD,
            nullptr, nullptr, nullptr, kB);

        // fused attention (Q=K=V=k)
        attn_kernel<<<dim3((NN + 15) / 16, HH, BB), 32, 0, stream>>>(kB, oB);

        // pre = o @ Wo[l] + x -> f32
        gemm_bf16_kernel<0, 1, 0, 1, 0><<<dim3(DD / 64, BN / 64), 128, 0, stream>>>(
            oB, woT + (size_t)l * DD * DD, BN, DD, DD,
            nullptr, xF, preF, nullptr);

        // z = LN(pre)
        layernorm_kernel<<<BN / 8, 256, 0, stream>>>(
            preF, gamma + (size_t)l * DD, beta + (size_t)l * DD, zF, zB, BN);

        // h = relu(z @ W1[l] + b1[l]) -> bf16
        gemm_bf16_kernel<1, 0, 1, 0, 1><<<dim3(DFF / 64, BN / 64), 128, 0, stream>>>(
            zB, w1T + (size_t)l * DFF * DD, BN, DFF, DD,
            b1 + (size_t)l * DFF, nullptr, nullptr, hB);

        // pre = h @ W2[l] + b2[l] + z -> f32
        gemm_bf16_kernel<1, 1, 0, 1, 0><<<dim3(DD / 64, BN / 64), 128, 0, stream>>>(
            hB, w2T + (size_t)l * DD * DFF, BN, DD, DFF,
            b2 + (size_t)l * DD, zF, preF, nullptr);

        // x = LN(pre)  (last layer writes straight to d_out)
        float* xOut = (l == LL - 1) ? (float*)d_out : xF;
        layernorm_kernel<<<BN / 8, 256, 0, stream>>>(
            preF, gamma + (size_t)l * DD, beta + (size_t)l * DD, xOut, xB, BN);
    }
    (void)n_in; (void)in_sizes; (void)out_size; (void)ws_size;
}
